// TAB_2018634629363
// MI455X (gfx1250) — compile-verified, matrix operand order checked by reading, not run on hardware
//
#include <hip/hip_runtime.h>
#include <math.h>
#include <stdint.h>

#define NTOK 16384   // H*W
#define CDIM 128
#define MLPD 256
#define NT   8
#define HEADS 4
#define DQ   32
#define GSZ  128
#define KW   256
#define BATCH 8

typedef __attribute__((ext_vector_type(16))) _Float16 v16h;
typedef __attribute__((ext_vector_type(8)))  _Float16 v8h;
typedef __attribute__((ext_vector_type(8)))  float    v8f;

// ---------------------------------------------------------------------------
// WMMA fragment fast path.  ISA 7.12.2 16-bit layouts: per lane, a fragment is
// two contiguous 8-half runs:  K = kbase + half*8 + {0..7}  and
// K = kbase + 16 + half*8 + {0..7}  (half = lane>=16).  With f16 tiles whose
// row stride is a multiple of 8 halves, each run is one 16-byte load
// (ds_load_b128 / global_load_b128).
// ---------------------------------------------------------------------------
__device__ __forceinline__ v16h frag_ld(const _Float16* tile, int ld, int rowbase,
                                        int kbase, int lane) {
  const int half = lane >> 4;
  const _Float16* p = tile + (size_t)(rowbase + (lane & 15)) * ld + kbase + half * 8;
  v8h lo = *(const v8h*)p;
  v8h hi = *(const v8h*)(p + 16);
  return __builtin_shufflevector(lo, hi, 0, 1, 2, 3, 4, 5, 6, 7,
                                 8, 9, 10, 11, 12, 13, 14, 15);
}

__device__ __forceinline__ v8f wmma16(v16h a, v16h b, v8f c) {
  return __builtin_amdgcn_wmma_f32_16x16x32_f16(false, a, false, b, (short)0, c, false, false);
}

__device__ __forceinline__ float gelu_f(float x) {
  return 0.5f * x * (1.0f + erff(x * 0.70710678118654752f));
}

// ---------------------------------------------------------------------------
// LN1 + cluster bucket assignment (argmax over xn @ mnorm^T).
// ---------------------------------------------------------------------------
__global__ void ln1_bucket_kernel(const float* __restrict__ x,
                                  const float* __restrict__ w,
                                  const float* __restrict__ bparm,
                                  const float* __restrict__ mnorm,
                                  float* __restrict__ xt, float* __restrict__ xn,
                                  int* __restrict__ bucket) {
  __shared__ float smn[NT * CDIM];
  for (int t = threadIdx.x; t < NT * CDIM; t += blockDim.x) smn[t] = mnorm[t];
  __syncthreads();
  int gidx = blockIdx.x * blockDim.x + threadIdx.x;
  int b = gidx / NTOK, n = gidx - b * NTOK;
  const float* xb = x + (size_t)b * CDIM * NTOK;
  float s = 0.f, ss = 0.f;
  for (int c = 0; c < CDIM; ++c) {
    float v = xb[(size_t)c * NTOK + n];
    s += v; ss += v * v;
  }
  float mean = s * (1.0f / CDIM);
  float var = ss * (1.0f / CDIM) - mean * mean;
  float rs = rsqrtf(var + 1e-5f);
  float dots[NT];
#pragma unroll
  for (int j = 0; j < NT; ++j) dots[j] = 0.f;
  size_t ob = ((size_t)b * NTOK + n) * CDIM;
  for (int c = 0; c < CDIM; ++c) {
    float v = xb[(size_t)c * NTOK + n];
    xt[ob + c] = v;
    float xv = (v - mean) * rs * w[c] + bparm[c];
    xn[ob + c] = xv;
#pragma unroll
    for (int j = 0; j < NT; ++j) dots[j] += xv * smn[j * CDIM + c];
  }
  int bk = 0; float best = dots[0];
#pragma unroll
  for (int j = 1; j < NT; ++j) if (dots[j] > best) { best = dots[j]; bk = j; }
  bucket[gidx] = bk;
}

// ---------------------------------------------------------------------------
// Precompute: normalized means, kg/vg, fused wcomb = conv1x1_w @ wproj.
// ---------------------------------------------------------------------------
__global__ void prep_kernel(const float* __restrict__ means,
                            const float* __restrict__ irca_wk,
                            const float* __restrict__ irca_wv,
                            const float* __restrict__ conv_w,
                            const float* __restrict__ wproj,
                            float* __restrict__ mnorm, float* __restrict__ kg,
                            float* __restrict__ vg, float* __restrict__ wcomb) {
  __shared__ float nrm[NT];
  if (threadIdx.x < NT) {
    float s = 0.f;
    for (int c = 0; c < CDIM; ++c) { float v = means[threadIdx.x * CDIM + c]; s += v * v; }
    nrm[threadIdx.x] = fmaxf(sqrtf(s), 1e-12f);
  }
  __syncthreads();
  for (int t = threadIdx.x; t < NT * CDIM; t += blockDim.x)
    mnorm[t] = means[t] / nrm[t / CDIM];
  for (int t = threadIdx.x; t < HEADS * NT * DQ; t += blockDim.x) {
    int h = t >> 8, rem = t & 255, nt = rem >> 5, d = rem & 31;
    float sk = 0.f, sv = 0.f;
    for (int c = 0; c < CDIM; ++c) {
      float m = means[nt * CDIM + c];
      sk += m * irca_wk[(h * DQ + d) * CDIM + c];
      sv += m * irca_wv[(h * DQ + d) * CDIM + c];
    }
    kg[t] = sk; vg[t] = sv;
  }
  for (int t = threadIdx.x; t < CDIM * CDIM; t += blockDim.x) {
    int o = t >> 7, c = t & 127;
    float s = 0.f;
    for (int k = 0; k < CDIM; ++k) s += conv_w[o * CDIM + k] * wproj[k * CDIM + c];
    wcomb[t] = s;
  }
}

// ---------------------------------------------------------------------------
// Stable counting sort by bucket (matches jnp.argsort stability).
// ---------------------------------------------------------------------------
__global__ void sort_kernel(const int* __restrict__ bucket, int* __restrict__ idx,
                            int* __restrict__ inv) {
  if (threadIdx.x != 0) return;
  int b = blockIdx.x;
  const int* bb = bucket + (size_t)b * NTOK;
  int cnt[NT];
#pragma unroll
  for (int j = 0; j < NT; ++j) cnt[j] = 0;
  for (int n = 0; n < NTOK; ++n) cnt[bb[n]]++;
  int ofs[NT]; int a = 0;
#pragma unroll
  for (int j = 0; j < NT; ++j) { ofs[j] = a; a += cnt[j]; }
  for (int n = 0; n < NTOK; ++n) {
    int r = ofs[bb[n]]++;
    idx[(size_t)b * NTOK + r] = n;
    inv[(size_t)b * NTOK + n] = r;
  }
}

// ---------------------------------------------------------------------------
// Templated WMMA GEMM:  Y = gather(X) @ W^T (+bias)(+add)(gelu)(NCHW store).
// Block = 4 waves, 64x128 output tile; wave = 32x64 (8 accumulators).
// A (64xK) and B (128xK) staged in LDS as f16; K-loop fully unrolled.
// grid(NTOK/64, Nout/128, B), block 128.
// ---------------------------------------------------------------------------
template <int K, typename TY, bool GATHER, bool HASBIAS, bool HASADD, bool ACT, bool TOUT>
__global__ void wmma_gemm_kernel(const float* __restrict__ X,
                                 const float* __restrict__ Wt,
                                 const float* __restrict__ bias,
                                 const float* __restrict__ addsrc,
                                 const int* __restrict__ gather,
                                 TY* __restrict__ Y, int Nout) {
  constexpr int KC = 128;
  const int b = blockIdx.z;
  const int rblk = blockIdx.x * 64;
  const int cblk = blockIdx.y * 128;
  const int wave = threadIdx.x >> 5, lane = threadIdx.x & 31;
  const int wr = wave >> 1, wc = wave & 1;
  __shared__ _Float16 sA[64 * KC];   // 16 KB
  __shared__ _Float16 sB[128 * KC];  // 32 KB
  __shared__ int srow[64];
  if (threadIdx.x < 64) {
    int row = rblk + threadIdx.x;
    srow[threadIdx.x] = GATHER ? gather[(size_t)b * NTOK + row] : row;
  }
  __syncthreads();

  v8f acc[2][4] = {};

#pragma unroll
  for (int kb = 0; kb < K; kb += KC) {
    if (kb) __syncthreads();
    for (int t = threadIdx.x; t < 64 * (KC / 2); t += blockDim.x) {
      int r = t / (KC / 2), kk = (t - r * (KC / 2)) * 2;
      const float2 xv = *(const float2*)&X[((size_t)b * NTOK + srow[r]) * K + kb + kk];
      sA[r * KC + kk] = (_Float16)xv.x;
      sA[r * KC + kk + 1] = (_Float16)xv.y;
    }
    for (int t = threadIdx.x; t < 128 * (KC / 2); t += blockDim.x) {
      int r = t / (KC / 2), kk = (t - r * (KC / 2)) * 2;
      const float2 wv = *(const float2*)&Wt[(size_t)(cblk + r) * K + kb + kk];
      sB[r * KC + kk] = (_Float16)wv.x;
      sB[r * KC + kk + 1] = (_Float16)wv.y;
    }
    __syncthreads();
#pragma unroll
    for (int ks = 0; ks < KC; ks += 32) {
      v16h a0 = frag_ld(sA, KC, wr * 32, ks, lane);
      v16h a1 = frag_ld(sA, KC, wr * 32 + 16, ks, lane);
      v16h b0 = frag_ld(sB, KC, wc * 64, ks, lane);
      v16h b1 = frag_ld(sB, KC, wc * 64 + 16, ks, lane);
      v16h b2 = frag_ld(sB, KC, wc * 64 + 32, ks, lane);
      v16h b3 = frag_ld(sB, KC, wc * 64 + 48, ks, lane);
      acc[0][0] = wmma16(a0, b0, acc[0][0]);
      acc[0][1] = wmma16(a0, b1, acc[0][1]);
      acc[0][2] = wmma16(a0, b2, acc[0][2]);
      acc[0][3] = wmma16(a0, b3, acc[0][3]);
      acc[1][0] = wmma16(a1, b0, acc[1][0]);
      acc[1][1] = wmma16(a1, b1, acc[1][1]);
      acc[1][2] = wmma16(a1, b2, acc[1][2]);
      acc[1][3] = wmma16(a1, b3, acc[1][3]);
    }
  }

  const int rb = rblk + wr * 32;
  const int cb = cblk + wc * 64;
#pragma unroll
  for (int nt2 = 0; nt2 < 4; ++nt2) {
    const int col = cb + nt2 * 16 + (lane & 15);
    const float bv = HASBIAS ? bias[col] : 0.f;
#pragma unroll
    for (int mt = 0; mt < 2; ++mt) {
#pragma unroll
      for (int r = 0; r < 8; ++r) {
        int m = r + ((lane >> 4) << 3);
        int row = rb + mt * 16 + m;
        float v = acc[mt][nt2][r] + bv;
        if (HASADD) v += addsrc[((size_t)b * NTOK + row) * Nout + col];
        if (ACT) v = gelu_f(v);
        if (TOUT) Y[((size_t)b * Nout + col) * NTOK + row] = (TY)v;
        else      Y[((size_t)b * NTOK + row) * Nout + col] = (TY)v;
      }
    }
  }
}

// ---------------------------------------------------------------------------
// Attention.  q/k/v arrive as f16 in sorted token order.  One block per
// (g,h,b); 8 waves x 16 query rows.  K-window staged into LDS with the
// CDNA5 async-to-LDS path (ASYNCcnt); V staged transposed (channel-major)
// so the P@V B-operand also uses the two-b128 fragment fast path.
// ---------------------------------------------------------------------------
__device__ __forceinline__ int win_row(int j, int g) {
  if (j < GSZ) return g * GSZ + j;
  if (g < (NTOK / GSZ) - 1) return (g + 1) * GSZ + (j - GSZ);
  return NTOK - 1 - (j - GSZ);
}

__global__ void attn_kernel(const _Float16* __restrict__ qs,
                            const _Float16* __restrict__ ks,
                            const _Float16* __restrict__ vs,
                            const float* __restrict__ kg,
                            const float* __restrict__ vg,
                            float* __restrict__ outs) {
  const int g = blockIdx.x, h = blockIdx.y, b = blockIdx.z;
  const int tid = threadIdx.x, wave = tid >> 5, lane = tid & 31;
  __shared__ _Float16 sk[KW * DQ];    // 16 KB, key-major
  __shared__ _Float16 svt[DQ * KW];   // 16 KB, channel-major (transposed V)
  __shared__ float    ssc[8][16 * 32];
  __shared__ _Float16 sp[8][16 * 32];
  __shared__ float    srmax[8][16], srsum[8][16];
  __shared__ float    sa2[8][16][NT];

  // --- stage K window via async global->LDS (16B per op, 4 per key row) ---
  {
    const unsigned ldsbase = (unsigned)(uintptr_t)(&sk[0]);
    for (int t = tid; t < KW * 4; t += blockDim.x) {
      int j = t >> 2, c8 = (t & 3) * 8;
      int tr = win_row(j, g);
      const _Float16* gp = ks + ((size_t)b * NTOK + tr) * CDIM + h * DQ + c8;
      unsigned lo = ldsbase + (unsigned)((j * DQ + c8) * sizeof(_Float16));
      asm volatile("global_load_async_to_lds_b128 %0, %1, off"
                   :: "v"(lo), "v"(gp) : "memory");
    }
    asm volatile("s_wait_asynccnt 0x0" ::: "memory");
  }
  // --- stage V transposed ---
  for (int t = tid; t < DQ * KW; t += blockDim.x) {
    int c = t / KW, j = t - c * KW;
    int tr = win_row(j, g);
    svt[c * KW + j] = vs[((size_t)b * NTOK + tr) * CDIM + h * DQ + c];
  }
  __syncthreads();

  const int rbase = g * GSZ + wave * 16;
  const _Float16* qbase = qs + ((size_t)b * NTOK + rbase) * CDIM + h * DQ;
  v16h aq = frag_ld(qbase, CDIM, 0, 0, lane);
  const float scale = 0.17677669529663687f;  // 1/sqrt(32)

  // ---- pass 1: running row max / sum over 256 local keys ----
  float rmax = -3.4e38f, rsum = 0.f;
  for (int kc = 0; kc < KW / 32; ++kc) {
#pragma unroll
    for (int t2 = 0; t2 < 2; ++t2) {
      v16h bm = frag_ld(sk, DQ, kc * 32 + t2 * 16, 0, lane);
      v8f s = {};
      s = wmma16(aq, bm, s);
#pragma unroll
      for (int r = 0; r < 8; ++r) {
        int m = r + ((lane >> 4) << 3);
        ssc[wave][m * 32 + t2 * 16 + (lane & 15)] = s[r] * scale;
      }
    }
    __syncthreads();
    if (lane < 16) {
      const float* srow = &ssc[wave][lane * 32];
      float cm = -3.4e38f;
      for (int j = 0; j < 32; ++j) cm = fmaxf(cm, srow[j]);
      float nm = fmaxf(rmax, cm);
      float acc = rsum * __expf(rmax - nm);
      for (int j = 0; j < 32; ++j) acc += __expf(srow[j] - nm);
      rmax = nm; rsum = acc;
    }
    __syncthreads();
  }
  if (lane < 16) { srmax[wave][lane] = rmax; srsum[wave][lane] = rsum; }

  // ---- global attention probs (8 cluster keys, VALU) ----
  if (lane < 16) {
    float s2[NT];
    float m2 = -3.4e38f;
#pragma unroll
    for (int j = 0; j < NT; ++j) {
      float d = 0.f;
      for (int c = 0; c < DQ; ++c)
        d += (float)qs[((size_t)b * NTOK + rbase + lane) * CDIM + h * DQ + c] *
             kg[(h * NT + j) * DQ + c];
      s2[j] = d * scale;
      m2 = fmaxf(m2, s2[j]);
    }
    float sum2 = 0.f;
#pragma unroll
    for (int j = 0; j < NT; ++j) { s2[j] = __expf(s2[j] - m2); sum2 += s2[j]; }
#pragma unroll
    for (int j = 0; j < NT; ++j) sa2[wave][lane][j] = s2[j] / sum2;
  }
  __syncthreads();

  // ---- pass 2: P @ V ----
  v8f o0 = {}, o1 = {};
  for (int kc = 0; kc < KW / 32; ++kc) {
#pragma unroll
    for (int t2 = 0; t2 < 2; ++t2) {
      v16h bm = frag_ld(sk, DQ, kc * 32 + t2 * 16, 0, lane);
      v8f s = {};
      s = wmma16(aq, bm, s);
#pragma unroll
      for (int r = 0; r < 8; ++r) {
        int m = r + ((lane >> 4) << 3);
        ssc[wave][m * 32 + t2 * 16 + (lane & 15)] = s[r] * scale;
      }
    }
    __syncthreads();
    if (lane < 16) {
      float rm = srmax[wave][lane], ri = 1.f / srsum[wave][lane];
      for (int j = 0; j < 32; ++j)
        sp[wave][lane * 32 + j] = (_Float16)(__expf(ssc[wave][lane * 32 + j] - rm) * ri);
    }
    __syncthreads();
    v16h ap = frag_ld(sp[wave], 32, 0, 0, lane);
    v16h bv0 = frag_ld(svt, KW, 0, kc * 32, lane);
    v16h bv1 = frag_ld(svt, KW, 16, kc * 32, lane);
    o0 = wmma16(ap, bv0, o0);
    o1 = wmma16(ap, bv1, o1);
    __syncthreads();
  }

  // ---- epilogue: add global-attention output, store (sorted order) ----
  const int c0 = lane & 15;
#pragma unroll
  for (int r = 0; r < 8; ++r) {
    int m = r + ((lane >> 4) << 3);
    float o2a = 0.f, o2b = 0.f;
#pragma unroll
    for (int j = 0; j < NT; ++j) {
      float p = sa2[wave][m][j];
      o2a += p * vg[(h * NT + j) * DQ + c0];
      o2b += p * vg[(h * NT + j) * DQ + 16 + c0];
    }
    size_t dst = ((size_t)b * NTOK + rbase + m) * CDIM + h * DQ;
    outs[dst + c0] = o0[r] + o2a;
    outs[dst + 16 + c0] = o1[r] + o2b;
  }
}

// ---------------------------------------------------------------------------
__global__ void ln2_kernel(const float* __restrict__ x2, const float* __restrict__ w,
                           const float* __restrict__ bparm, float* __restrict__ xm) {
  int gidx = blockIdx.x * blockDim.x + threadIdx.x;
  const float* row = x2 + (size_t)gidx * CDIM;
  float s = 0.f, ss = 0.f;
  for (int c = 0; c < CDIM; ++c) { float v = row[c]; s += v; ss += v * v; }
  float mean = s * (1.0f / CDIM);
  float var = ss * (1.0f / CDIM) - mean * mean;
  float rs = rsqrtf(var + 1e-5f);
  float* orow = xm + (size_t)gidx * CDIM;
  for (int c = 0; c < CDIM; ++c)
    orow[c] = (row[c] - mean) * rs * w[c] + bparm[c];
}

// ---------------------------------------------------------------------------
__global__ void dwconv_kernel(const float* __restrict__ h1, const float* __restrict__ w,
                              const float* __restrict__ bias, float* __restrict__ h1b) {
  int ch = threadIdx.x;
  int nf = blockIdx.x;
  int b = nf / NTOK, n = nf - b * NTOK;
  int y = n >> 7, x0 = n & 127;
  float acc = 0.f;
#pragma unroll
  for (int dy = 0; dy < 5; ++dy) {
    int yy = y + dy - 2;
    if (yy < 0 || yy >= 128) continue;
#pragma unroll
    for (int dx = 0; dx < 5; ++dx) {
      int xx = x0 + dx - 2;
      if (xx < 0 || xx >= 128) continue;
      acc += w[ch * 25 + dy * 5 + dx] *
             h1[((size_t)b * NTOK + yy * 128 + xx) * MLPD + ch];
    }
  }
  float gv = gelu_f(acc + bias[ch]);
  size_t i = ((size_t)b * NTOK + n) * MLPD + ch;
  h1b[i] = h1[i] + gv;
}

// ---------------------------------------------------------------------------
extern "C" void kernel_launch(void* const* d_in, const int* in_sizes, int n_in,
                              void* d_out, int out_size, void* d_ws, size_t ws_size,
                              hipStream_t stream) {
  const float* x        = (const float*)d_in[0];
  const float* means    = (const float*)d_in[1];
  const float* ln1_w    = (const float*)d_in[2];
  const float* ln1_b    = (const float*)d_in[3];
  const float* wq       = (const float*)d_in[4];
  const float* wk       = (const float*)d_in[5];
  const float* wv       = (const float*)d_in[6];
  const float* wproj    = (const float*)d_in[7];
  const float* irca_wk  = (const float*)d_in[8];
  const float* irca_wv  = (const float*)d_in[9];
  const float* conv1x1  = (const float*)d_in[10];
  const float* ln2_w    = (const float*)d_in[11];
  const float* ln2_b    = (const float*)d_in[12];
  const float* fc1_w    = (const float*)d_in[13];
  const float* fc1_b    = (const float*)d_in[14];
  const float* dw_w     = (const float*)d_in[15];
  const float* dw_b     = (const float*)d_in[16];
  const float* fc2_w    = (const float*)d_in[17];
  const float* fc2_b    = (const float*)d_in[18];
  float* out = (float*)d_out;

  const size_t BN = (size_t)BATCH * NTOK;
  char* p = (char*)d_ws;
  auto carve = [&](size_t bytes) {
    void* r = (void*)p;
    p += (bytes + 255) & ~(size_t)255;
    return r;
  };
  float* xt    = (float*)carve(BN * CDIM * 4);
  float* xn    = (float*)carve(BN * CDIM * 4);
  float* x2    = (float*)carve(BN * CDIM * 4);
  float* xm    = (float*)carve(BN * CDIM * 4);
  _Float16* qsb = (_Float16*)carve(BN * CDIM * 2);
  _Float16* ksb = (_Float16*)carve(BN * CDIM * 2);
  _Float16* vsb = (_Float16*)carve(BN * CDIM * 2);
  float* outs  = (float*)carve(BN * CDIM * 4);
  float* h1    = (float*)carve(BN * MLPD * 4);
  float* h1b   = (float*)carve(BN * MLPD * 4);
  float* mnorm = (float*)carve(NT * CDIM * 4);
  float* kg    = (float*)carve(HEADS * NT * DQ * 4);
  float* vg    = (float*)carve(HEADS * NT * DQ * 4);
  float* wcomb = (float*)carve(CDIM * CDIM * 4);
  int* bucket  = (int*)carve(BN * 4);
  int* idx     = (int*)carve(BN * 4);
  int* inv     = (int*)carve(BN * 4);

  prep_kernel<<<1, 256, 0, stream>>>(means, irca_wk, irca_wv, conv1x1, wproj,
                                     mnorm, kg, vg, wcomb);
  ln1_bucket_kernel<<<BN / 256, 256, 0, stream>>>(x, ln1_w, ln1_b, mnorm, xt, xn, bucket);
  sort_kernel<<<BATCH, 32, 0, stream>>>(bucket, idx, inv);

  dim3 gq(NTOK / 64, CDIM / 128, BATCH);   // (256,1,8)
  dim3 gm(NTOK / 64, MLPD / 128, BATCH);   // (256,2,8)
  // gathered q/k/v projections, stored f16 in sorted token order
  wmma_gemm_kernel<CDIM, _Float16, true, false, false, false, false>
      <<<gq, 128, 0, stream>>>(xn, wq, nullptr, nullptr, idx, qsb, CDIM);
  wmma_gemm_kernel<CDIM, _Float16, true, false, false, false, false>
      <<<gq, 128, 0, stream>>>(xn, wk, nullptr, nullptr, idx, ksb, CDIM);
  wmma_gemm_kernel<CDIM, _Float16, true, false, false, false, false>
      <<<gq, 128, 0, stream>>>(xn, wv, nullptr, nullptr, idx, vsb, CDIM);

  dim3 ga(NTOK / GSZ, HEADS, BATCH);
  attn_kernel<<<ga, 256, 0, stream>>>(qsb, ksb, vsb, kg, vg, outs);

  // scatter-back (inverse perm) + fused wproj/conv1x1 + residual
  wmma_gemm_kernel<CDIM, float, true, false, true, false, false>
      <<<gq, 128, 0, stream>>>(outs, wcomb, nullptr, xt, inv, x2, CDIM);
  ln2_kernel<<<BN / 256, 256, 0, stream>>>(x2, ln2_w, ln2_b, xm);
  // fc1 + gelu
  wmma_gemm_kernel<CDIM, float, false, true, false, true, false>
      <<<gm, 128, 0, stream>>>(xm, fc1_w, fc1_b, nullptr, nullptr, h1, MLPD);
  dwconv_kernel<<<BN, MLPD, 0, stream>>>(h1, dw_w, dw_b, h1b);
  // fc2 + residual, stored directly as NCHW output
  wmma_gemm_kernel<MLPD, float, false, true, true, false, true>
      <<<gq, 128, 0, stream>>>(h1b, fc2_w, fc2_b, x2, nullptr, out, CDIM);
}
